// GNNEncoder_62302795596683
// MI455X (gfx1250) — matrix operand order repaired and checked by reading, not compiled, not run on hardware
//
#include <hip/hip_runtime.h>
#include <hip/hip_bf16.h>

typedef __attribute__((ext_vector_type(16))) _Float16 v16h;
typedef __attribute__((ext_vector_type(8)))  float    v8f;

#define HD     64
#define LAYERS 5
#define NIDIM  92

#define SZ_NODE (3*4*512)   // 3 k-chunks x 4 n-tiles x 32 lanes x 16 halves
#define SZ_W1   (4*4*512)
#define SZ_W2   (2*4*512)
#define SZ_UPD  (4*4*512)

__device__ __forceinline__ v8f wmma16(v16h a, v16h b, v8f c) {
  return __builtin_amdgcn_wmma_f32_16x16x32_f16(false, a, false, b, (short)0, c, false, false);
}
__device__ __forceinline__ float silu_f(float x) { return x / (1.0f + __expf(-x)); }

// ---------------- utility: zero f32 ----------------
__global__ void k_zero(float* __restrict__ p, long long n) {
  long long i = (long long)blockIdx.x * blockDim.x + threadIdx.x;
  long long s = (long long)gridDim.x * blockDim.x;
  for (; i < n; i += s) p[i] = 0.0f;
}

// ---------------- weight prepack into B-fragment order ----------------
// B-fragment layout (16-bit 32x16): lane l, half h -> K = kc*32 + 16*(h>>3) + 8*(l>>4) + (h&7),
//                                                     N = nt*16 + (l&15)
__device__ __forceinline__ void pack_one(_Float16* dst, const float* W, int Kreal, int idx) {
  int hh   = idx & 15;
  int lane = (idx >> 4) & 31;
  int frag = idx >> 9;            // kc*4 + nt
  int kc = frag >> 2, nt = frag & 3;
  int k = kc * 32 + ((hh >> 3) << 4) + ((lane >> 4) << 3) + (hh & 7);
  int n = nt * 16 + (lane & 15);
  float v = (k < Kreal) ? W[(size_t)k * 64 + n] : 0.0f;
  dst[idx] = (_Float16)v;
}

__global__ void k_prepack(const float* __restrict__ node_W, const float* __restrict__ msg_W1,
                          const float* __restrict__ msg_W2, const float* __restrict__ upd_W,
                          _Float16* __restrict__ pnode, _Float16* __restrict__ pw1,
                          _Float16* __restrict__ pw2, _Float16* __restrict__ pupd,
                          float* __restrict__ w1last) {
  int i = blockIdx.x * blockDim.x + threadIdx.x;
  int s = gridDim.x * blockDim.x;
  const int total = SZ_NODE + LAYERS * (SZ_W1 + SZ_W2 + SZ_UPD) + LAYERS * 64;
  for (; i < total; i += s) {
    int r = i;
    if (r < SZ_NODE) { pack_one(pnode, node_W, NIDIM, r); continue; }
    r -= SZ_NODE;
    if (r < LAYERS * SZ_W1) {
      int l = r / SZ_W1;
      pack_one(pw1 + (size_t)l * SZ_W1, msg_W1 + (size_t)l * 129 * 64, 128, r % SZ_W1);
      continue;
    }
    r -= LAYERS * SZ_W1;
    if (r < LAYERS * SZ_W2) {
      int l = r / SZ_W2;
      pack_one(pw2 + (size_t)l * SZ_W2, msg_W2 + (size_t)l * 64 * 64, 64, r % SZ_W2);
      continue;
    }
    r -= LAYERS * SZ_W2;
    if (r < LAYERS * SZ_UPD) {
      int l = r / SZ_UPD;
      pack_one(pupd + (size_t)l * SZ_UPD, upd_W + (size_t)l * 128 * 64, 128, r % SZ_UPD);
      continue;
    }
    r -= LAYERS * SZ_UPD;
    { int l = r / 64, f = r % 64; w1last[l * 64 + f] = msg_W1[(size_t)l * 129 * 64 + 128 * 64 + f]; }
  }
}

// ---------------- degree ----------------
__global__ void k_deg(const int* __restrict__ dstI, float* __restrict__ deg, int E) {
  int i = blockIdx.x * blockDim.x + threadIdx.x;
  int s = gridDim.x * blockDim.x;
  for (; i < E; i += s) atomicAdd(&deg[dstI[i]], 1.0f);
}

// ---------------- node embedding: h = x @ node_W + node_b ----------------
__global__ void __launch_bounds__(256, 1)
k_embed(const float* __restrict__ x, const _Float16* __restrict__ pnode,
        const float* __restrict__ nb, float* __restrict__ hbuf,
        _Float16* __restrict__ h16, int N) {
  int wave = threadIdx.x >> 5, lane = threadIdx.x & 31;
  int wid = blockIdx.x * (blockDim.x >> 5) + wave;
  int nw  = gridDim.x * (blockDim.x >> 5);
  int ntiles = (N + 15) >> 4;
  int g = lane >> 4;
  const v16h* pb = (const v16h*)pnode;
  v16h bf[12];                     // 3 kc x 4 nt, register-resident
#pragma unroll
  for (int q = 0; q < 12; ++q) bf[q] = pb[q * 32 + lane];
  float bias4[4];
#pragma unroll
  for (int nt = 0; nt < 4; ++nt) bias4[nt] = nb[nt * 16 + (lane & 15)];

  for (int tile = wid; tile < ntiles; tile += nw) {
    int n0 = tile * 16;
    bool full = (n0 + 16 <= N);
    int row = n0 + (lane & 15);
    int rowc = row < N ? row : N - 1;
    const float* xr = x + (size_t)rowc * NIDIM;
    v16h a[3];
#pragma unroll
    for (int kc = 0; kc < 3; ++kc)
#pragma unroll
      for (int hh = 0; hh < 16; ++hh) {
        int k = kc * 32 + ((hh >> 3) << 4) + (g << 3) + (hh & 7);   // A-frag K mapping
        a[kc][hh] = (_Float16)(k < NIDIM ? xr[k] : 0.0f);
      }
#pragma unroll
    for (int nt = 0; nt < 4; ++nt) {
      int f = nt * 16 + (lane & 15);
      v8f c = {};                              // SRC2 = inline 0; bias added in epilogue
#pragma unroll
      for (int kc = 0; kc < 3; ++kc) c = wmma16(a[kc], bf[kc * 4 + nt], c);
      if (full) {
#pragma unroll
        for (int r = 0; r < 8; ++r) {
          int nn = n0 + r + 8 * g;
          float v = c[r] + bias4[nt];
          hbuf[(size_t)nn * 64 + f] = v;
          h16[(size_t)nn * 64 + f]  = (_Float16)v;
        }
      } else {
#pragma unroll
        for (int r = 0; r < 8; ++r) {
          int nn = n0 + r + 8 * g;
          if (nn < N) {
            float v = c[r] + bias4[nt];
            hbuf[(size_t)nn * 64 + f] = v;
            h16[(size_t)nn * 64 + f]  = (_Float16)v;
          }
        }
      }
    }
  }
}

// ---------------- message pass 1: z = [h_dst, h_src, ea] @ W1 + b1 (pre-BN) ----------------
__global__ void __launch_bounds__(256, 1)
k_msg1(const _Float16* __restrict__ h16, const float* __restrict__ eattr,
       const int* __restrict__ srcI, const int* __restrict__ dstI,
       const _Float16* __restrict__ pw1, const float* __restrict__ w1last,
       const float* __restrict__ b1, _Float16* __restrict__ zbuf,
       float* __restrict__ stats, int E) {
  __shared__ float sstat[128];
  if (threadIdx.x < 128) sstat[threadIdx.x] = 0.0f;
  __syncthreads();
  int wave = threadIdx.x >> 5, lane = threadIdx.x & 31;
  int wid = blockIdx.x * (blockDim.x >> 5) + wave;
  int nw  = gridDim.x * (blockDim.x >> 5);
  int ntiles = (E + 15) >> 4;
  int g = lane >> 4;
  const v16h* pb = (const v16h*)pw1;
  v16h bf[16];                     // 4 kc x 4 nt, register-resident for wave lifetime
#pragma unroll
  for (int q = 0; q < 16; ++q) bf[q] = pb[q * 32 + lane];
  float bias4[4], wlf4[4];
#pragma unroll
  for (int nt = 0; nt < 4; ++nt) {
    int f = nt * 16 + (lane & 15);
    bias4[nt] = b1[f];
    wlf4[nt]  = w1last[f];
  }
  float sA[4] = {0, 0, 0, 0}, s2A[4] = {0, 0, 0, 0};  // per-feature stats in registers

  for (int tile = wid; tile < ntiles; tile += nw) {
    int e0 = tile * 16;
    bool full = (e0 + 16 <= E);
    int er = e0 + (lane & 15);
    int erc = er < E ? er : E - 1;
    const _Float16* rd = h16 + (size_t)dstI[erc] * 64;   // x_i = h[dst]
    const _Float16* rs = h16 + (size_t)srcI[erc] * 64;   // x_j = h[src]
    v16h a[4];
#pragma unroll
    for (int c2 = 0; c2 < 2; ++c2)
#pragma unroll
      for (int hh = 0; hh < 16; ++hh) {
        int k = c2 * 32 + ((hh >> 3) << 4) + (g << 3) + (hh & 7);
        a[c2][hh]     = rd[k];      // K 0..63   <- h[dst]
        a[2 + c2][hh] = rs[k];      // K 64..127 <- h[src]
      }
    float ea[8];
    if (full) {
#pragma unroll
      for (int r = 0; r < 8; ++r) ea[r] = eattr[e0 + r + 8 * g];
    } else {
#pragma unroll
      for (int r = 0; r < 8; ++r) {
        int e = e0 + r + 8 * g;
        ea[r] = (e < E) ? eattr[e] : 0.0f;
      }
    }
#pragma unroll
    for (int nt = 0; nt < 4; ++nt) {
      int f = nt * 16 + (lane & 15);
      v8f c = {};
#pragma unroll
      for (int kc = 0; kc < 4; ++kc) c = wmma16(a[kc], bf[kc * 4 + nt], c);
      if (full) {
#pragma unroll
        for (int r = 0; r < 8; ++r) {
          float z = c[r] + fmaf(ea[r], wlf4[nt], bias4[nt]);
          zbuf[(size_t)(e0 + r + 8 * g) * 64 + f] = (_Float16)z;
          sA[nt] += z; s2A[nt] += z * z;
        }
      } else {
#pragma unroll
        for (int r = 0; r < 8; ++r) {
          int e = e0 + r + 8 * g;
          float z = c[r] + fmaf(ea[r], wlf4[nt], bias4[nt]);
          if (e < E) {
            zbuf[(size_t)e * 64 + f] = (_Float16)z;
            sA[nt] += z; s2A[nt] += z * z;
          }
        }
      }
    }
  }
#pragma unroll
  for (int nt = 0; nt < 4; ++nt) {
    int f = nt * 16 + (lane & 15);
    atomicAdd(&sstat[f], sA[nt]);
    atomicAdd(&sstat[64 + f], s2A[nt]);
  }
  __syncthreads();
  if (threadIdx.x < 128) atomicAdd(&stats[threadIdx.x], sstat[threadIdx.x]);
}

// ---------------- BN finalize: fold into affine a*z + b ----------------
__global__ void k_bnfin(const float* __restrict__ stats, const float* __restrict__ gamma,
                        const float* __restrict__ beta, float cnt, float* __restrict__ ab) {
  int f = threadIdx.x;
  float mu  = stats[f] / cnt;
  float var = stats[64 + f] / cnt - mu * mu;
  var = fmaxf(var, 0.0f);
  float inv = rsqrtf(var + 1e-5f);
  float aa = gamma[f] * inv;
  ab[f] = aa;
  ab[64 + f] = beta[f] - mu * aa;
}

// ---------------- message pass 2: m = SiLU(BN(z)) @ W2 + b2, scatter-add ----------------
__global__ void __launch_bounds__(256, 1)
k_msg2(const _Float16* __restrict__ zbuf, const float* __restrict__ ab,
       const _Float16* __restrict__ pw2, const float* __restrict__ b2,
       const int* __restrict__ dstI, float* __restrict__ agg, int E) {
  __shared__ float sab[128];
  if (threadIdx.x < 128) sab[threadIdx.x] = ab[threadIdx.x];
  __syncthreads();
  int wave = threadIdx.x >> 5, lane = threadIdx.x & 31;
  int wid = blockIdx.x * (blockDim.x >> 5) + wave;
  int nw  = gridDim.x * (blockDim.x >> 5);
  int ntiles = (E + 15) >> 4;
  int g = lane >> 4;
  const v16h* pb = (const v16h*)pw2;
  v16h bf[8];                      // 2 kc x 4 nt
#pragma unroll
  for (int q = 0; q < 8; ++q) bf[q] = pb[q * 32 + lane];
  float bias4[4];
#pragma unroll
  for (int nt = 0; nt < 4; ++nt) bias4[nt] = b2[nt * 16 + (lane & 15)];

  for (int tile = wid; tile < ntiles; tile += nw) {
    int e0 = tile * 16;
    bool full = (e0 + 16 <= E);
    int er = e0 + (lane & 15);
    int erc = er < E ? er : E - 1;
    const _Float16* zr = zbuf + (size_t)erc * 64;
    v16h a[2];
#pragma unroll
    for (int c2 = 0; c2 < 2; ++c2)
#pragma unroll
      for (int hh = 0; hh < 16; ++hh) {
        int k = c2 * 32 + ((hh >> 3) << 4) + (g << 3) + (hh & 7);
        float z = (float)zr[k];
        z = sab[k] * z + sab[64 + k];
        a[c2][hh] = (_Float16)silu_f(z);
      }
    int d8[8];
#pragma unroll
    for (int r = 0; r < 8; ++r) {
      int e = e0 + r + 8 * g;
      d8[r] = (e < E) ? dstI[e] : 0;
    }
#pragma unroll
    for (int nt = 0; nt < 4; ++nt) {
      int f = nt * 16 + (lane & 15);
      v8f c = {};
#pragma unroll
      for (int kc = 0; kc < 2; ++kc) c = wmma16(a[kc], bf[kc * 4 + nt], c);
      if (full) {
#pragma unroll
        for (int r = 0; r < 8; ++r)
          atomicAdd(&agg[(size_t)d8[r] * 64 + f], c[r] + bias4[nt]);
      } else {
#pragma unroll
        for (int r = 0; r < 8; ++r)
          if (e0 + r + 8 * g < E)
            atomicAdd(&agg[(size_t)d8[r] * 64 + f], c[r] + bias4[nt]);
      }
    }
  }
}

// ---------------- update pass 1: u = [h, agg/deg] @ upd_W + b (pre-BN) ----------------
__global__ void __launch_bounds__(256, 1)
k_upd1(const _Float16* __restrict__ h16, const float* __restrict__ agg,
       const float* __restrict__ deg, const _Float16* __restrict__ pupd,
       const float* __restrict__ ub, float* __restrict__ ubuf,
       float* __restrict__ stats, int N) {
  __shared__ float sstat[128];
  if (threadIdx.x < 128) sstat[threadIdx.x] = 0.0f;
  __syncthreads();
  int wave = threadIdx.x >> 5, lane = threadIdx.x & 31;
  int wid = blockIdx.x * (blockDim.x >> 5) + wave;
  int nw  = gridDim.x * (blockDim.x >> 5);
  int ntiles = (N + 15) >> 4;
  int g = lane >> 4;
  const v16h* pb = (const v16h*)pupd;
  v16h bf[16];
#pragma unroll
  for (int q = 0; q < 16; ++q) bf[q] = pb[q * 32 + lane];
  float bias4[4];
#pragma unroll
  for (int nt = 0; nt < 4; ++nt) bias4[nt] = ub[nt * 16 + (lane & 15)];
  float sA[4] = {0, 0, 0, 0}, s2A[4] = {0, 0, 0, 0};

  for (int tile = wid; tile < ntiles; tile += nw) {
    int n0 = tile * 16;
    bool full = (n0 + 16 <= N);
    int row = n0 + (lane & 15);
    int rowc = row < N ? row : N - 1;
    const _Float16* hr = h16 + (size_t)rowc * 64;
    const float* ar = agg + (size_t)rowc * 64;
    float rdeg = 1.0f / fmaxf(deg[rowc], 1.0f);
    v16h a[4];
#pragma unroll
    for (int c2 = 0; c2 < 2; ++c2)
#pragma unroll
      for (int hh = 0; hh < 16; ++hh) {
        int k = c2 * 32 + ((hh >> 3) << 4) + (g << 3) + (hh & 7);
        a[c2][hh]     = hr[k];                       // K 0..63   <- h
        a[2 + c2][hh] = (_Float16)(ar[k] * rdeg);    // K 64..127 <- mean agg
      }
#pragma unroll
    for (int nt = 0; nt < 4; ++nt) {
      int f = nt * 16 + (lane & 15);
      v8f c = {};
#pragma unroll
      for (int kc = 0; kc < 4; ++kc) c = wmma16(a[kc], bf[kc * 4 + nt], c);
      if (full) {
#pragma unroll
        for (int r = 0; r < 8; ++r) {
          int nn = n0 + r + 8 * g;
          float v = c[r] + bias4[nt];
          ubuf[(size_t)nn * 64 + f] = v;
          sA[nt] += v; s2A[nt] += v * v;
        }
      } else {
#pragma unroll
        for (int r = 0; r < 8; ++r) {
          int nn = n0 + r + 8 * g;
          if (nn < N) {
            float v = c[r] + bias4[nt];
            ubuf[(size_t)nn * 64 + f] = v;
            sA[nt] += v; s2A[nt] += v * v;
          }
        }
      }
    }
  }
#pragma unroll
  for (int nt = 0; nt < 4; ++nt) {
    int f = nt * 16 + (lane & 15);
    atomicAdd(&sstat[f], sA[nt]);
    atomicAdd(&sstat[64 + f], s2A[nt]);
  }
  __syncthreads();
  if (threadIdx.x < 128) atomicAdd(&stats[threadIdx.x], sstat[threadIdx.x]);
}

// ---------------- update pass 2: h = SiLU(BN(u)) ----------------
__global__ void k_upd2(const float* __restrict__ ubuf, const float* __restrict__ ab,
                       float* __restrict__ hbuf, _Float16* __restrict__ h16, long long total) {
  long long i = (long long)blockIdx.x * blockDim.x + threadIdx.x;
  long long s = (long long)gridDim.x * blockDim.x;
  for (; i < total; i += s) {
    int f = (int)(i & 63);
    float t = ab[f] * ubuf[i] + ab[64 + f];
    float hn = silu_f(t);
    hbuf[i] = hn;
    h16[i] = (_Float16)hn;
  }
}

// ---------------- global mean pool ----------------
__global__ void k_pool(const float* __restrict__ hbuf, const int* __restrict__ batch,
                       float* __restrict__ gsum, float* __restrict__ gcnt, long long total) {
  long long i = (long long)blockIdx.x * blockDim.x + threadIdx.x;
  long long s = (long long)gridDim.x * blockDim.x;
  for (; i < total; i += s) {
    int n = (int)(i >> 6), f = (int)(i & 63);
    int b = batch[n];
    atomicAdd(&gsum[(size_t)b * 64 + f], hbuf[i]);
    if (f == 0) atomicAdd(&gcnt[b], 1.0f);
  }
}

// ---------------- output projection: out = SiLU(g @ out_W + out_b) ----------------
__global__ void k_out(const float* __restrict__ gsum, const float* __restrict__ gcnt,
                      const float* __restrict__ oW, const float* __restrict__ ob,
                      float* __restrict__ out) {
  __shared__ float gm[64];
  int gi = blockIdx.x;
  int j = threadIdx.x;
  float cnt = fmaxf(gcnt[gi], 1.0f);
  gm[j] = gsum[(size_t)gi * 64 + j] / cnt;
  __syncthreads();
  float acc = ob[j];
#pragma unroll 8
  for (int k = 0; k < 64; ++k) acc += gm[k] * oW[k * 64 + j];
  out[(size_t)gi * 64 + j] = silu_f(acc);
}

extern "C" void kernel_launch(void* const* d_in, const int* in_sizes, int n_in,
                              void* d_out, int out_size, void* d_ws, size_t ws_size,
                              hipStream_t stream) {
  (void)n_in; (void)ws_size;
  const float* x       = (const float*)d_in[0];
  const float* eattr   = (const float*)d_in[1];
  const int*   ei      = (const int*)d_in[2];
  const int*   batch   = (const int*)d_in[3];
  const float* node_W  = (const float*)d_in[4];
  const float* node_b  = (const float*)d_in[5];
  const float* msg_W1  = (const float*)d_in[6];
  const float* msg_b1  = (const float*)d_in[7];
  const float* msg_g1  = (const float*)d_in[8];
  const float* msg_be1 = (const float*)d_in[9];
  const float* msg_W2  = (const float*)d_in[10];
  const float* msg_b2  = (const float*)d_in[11];
  const float* upd_W   = (const float*)d_in[12];
  const float* upd_b   = (const float*)d_in[13];
  const float* upd_g   = (const float*)d_in[14];
  const float* upd_be  = (const float*)d_in[15];
  const float* out_W   = (const float*)d_in[16];
  const float* out_b   = (const float*)d_in[17];

  const int N = in_sizes[0] / NIDIM;
  const int E = in_sizes[1];
  const int G = out_size / 64;
  const int* srcI = ei;
  const int* dstI = ei + E;

  char* w = (char*)d_ws;
  auto carve = [&](size_t bytes) -> char* {
    char* p = w;
    w += (bytes + 255) & ~(size_t)255;
    return p;
  };
  float*    hbuf   = (float*)carve((size_t)N * 64 * 4);
  _Float16* h16    = (_Float16*)carve((size_t)N * 64 * 2);
  _Float16* zbuf   = (_Float16*)carve((size_t)E * 64 * 2);
  float*    ubuf   = (float*)carve((size_t)N * 64 * 4);
  float*    agg    = (float*)carve((size_t)N * 64 * 4);
  float*    deg    = (float*)carve((size_t)N * 4);
  float*    stats  = (float*)carve(128 * 4);
  float*    ab     = (float*)carve(128 * 4);
  float*    gsum   = (float*)carve((size_t)G * 64 * 4);
  float*    gcnt   = (float*)carve((size_t)G * 4);
  _Float16* pnode  = (_Float16*)carve((size_t)SZ_NODE * 2);
  _Float16* pw1    = (_Float16*)carve((size_t)LAYERS * SZ_W1 * 2);
  _Float16* pw2    = (_Float16*)carve((size_t)LAYERS * SZ_W2 * 2);
  _Float16* pupd   = (_Float16*)carve((size_t)LAYERS * SZ_UPD * 2);
  float*    w1last = (float*)carve((size_t)LAYERS * 64 * 4);

  const int ntile = (N + 15) / 16;
  const int etile = (E + 15) / 16;
  auto capmin = [](int a, int b) { return a < b ? a : b; };
  const int nblk = capmin((ntile + 7) / 8, 512);    // persistent waves
  const int eblk = capmin((etile + 7) / 8, 1536);   // persistent waves

  k_prepack<<<128, 256, 0, stream>>>(node_W, msg_W1, msg_W2, upd_W, pnode, pw1, pw2, pupd, w1last);
  k_zero<<<256, 256, 0, stream>>>(deg, N);
  k_deg<<<1024, 256, 0, stream>>>(dstI, deg, E);
  k_embed<<<nblk, 256, 0, stream>>>(x, pnode, node_b, hbuf, h16, N);

  for (int l = 0; l < LAYERS; ++l) {
    k_zero<<<1, 128, 0, stream>>>(stats, 128);
    k_msg1<<<eblk, 256, 0, stream>>>(h16, eattr, srcI, dstI, pw1 + (size_t)l * SZ_W1,
                                     w1last + l * 64, msg_b1 + l * 64, zbuf, stats, E);
    k_bnfin<<<1, 64, 0, stream>>>(stats, msg_g1 + l * 64, msg_be1 + l * 64, (float)E, ab);
    k_zero<<<512, 256, 0, stream>>>(agg, (long long)N * 64);
    k_msg2<<<eblk, 256, 0, stream>>>(zbuf, ab, pw2 + (size_t)l * SZ_W2, msg_b2 + l * 64,
                                     dstI, agg, E);
    k_zero<<<1, 128, 0, stream>>>(stats, 128);
    k_upd1<<<nblk, 256, 0, stream>>>(h16, agg, deg, pupd + (size_t)l * SZ_UPD,
                                     upd_b + l * 64, ubuf, stats, N);
    k_bnfin<<<1, 64, 0, stream>>>(stats, upd_g + l * 64, upd_be + l * 64, (float)N, ab);
    k_upd2<<<512, 256, 0, stream>>>(ubuf, ab, hbuf, h16, (long long)N * 64);
  }

  k_zero<<<64, 256, 0, stream>>>(gsum, (long long)G * 64);
  k_zero<<<1, 256, 0, stream>>>(gcnt, G);
  k_pool<<<512, 256, 0, stream>>>(hbuf, batch, gsum, gcnt, (long long)N * 64);
  k_out<<<G, 64, 0, stream>>>(gsum, gcnt, out_W, out_b, (float*)d_out);
}
